// DN4_67035849556464
// MI455X (gfx1250) — matrix operand order repaired and checked by reading, not compiled, not run on hardware
//
#include <hip/hip_runtime.h>
#include <hip/hip_bf16.h>

typedef __attribute__((ext_vector_type(16))) _Float16 v16h;
typedef __attribute__((ext_vector_type(8)))  _Float16 v8h;
typedef __attribute__((ext_vector_type(8)))  float    v8f;

#define BATCH 2
#define NWAY  5
#define KSHOT 5
#define NQ    75
#define CH    64
#define HW    441
#define HWP   448          // padded to multiple of 16
#define NT    (HWP / 16)   // 28 tiles of 16

// ---------------------------------------------------------------------------
// Kernel 1: support -> shot-mean -> L2-normalize(channel) -> f16, layout
// Bt[(b*NWAY+w)][n][c], rows 441..447 zeroed.
// ---------------------------------------------------------------------------
__global__ void dn4_prep_support(const float* __restrict__ sup,
                                 _Float16* __restrict__ Bt) {
  int idx = blockIdx.x * blockDim.x + threadIdx.x;
  const int total = BATCH * NWAY * HWP;
  if (idx >= total) return;
  int n = idx % HWP;
  int t = idx / HWP;
  int w = t % NWAY;
  int b = t / NWAY;
  _Float16* dst = Bt + ((size_t)(b * NWAY + w) * HWP + n) * CH;
  if (n >= HW) {
#pragma unroll
    for (int cc = 0; cc < CH; ++cc) dst[cc] = (_Float16)0.0f;
    return;
  }
  float v[CH];
  float ss = 0.0f;
  const float inv = 1.0f / (float)KSHOT;
#pragma unroll 4
  for (int cc = 0; cc < CH; ++cc) {
    float acc = 0.0f;
#pragma unroll
    for (int s = 0; s < KSHOT; ++s) {
      acc += sup[(((size_t)b * (NWAY * KSHOT) + w * KSHOT + s) * CH + cc) * HW + n];
    }
    acc *= inv;
    v[cc] = acc;
    ss += acc * acc;
  }
  float r = 1.0f / fmaxf(sqrtf(ss), 1e-12f);
#pragma unroll 4
  for (int cc = 0; cc < CH; ++cc) dst[cc] = (_Float16)(v[cc] * r);
}

// ---------------------------------------------------------------------------
// Kernel 2: query -> L2-normalize(channel) -> f16, layout Qf[(b*NQ+q)][m][c],
// rows 441..447 zeroed (padded rows then contribute exactly 0 to the sum).
// ---------------------------------------------------------------------------
__global__ void dn4_prep_query(const float* __restrict__ qry,
                               _Float16* __restrict__ Qf) {
  int idx = blockIdx.x * blockDim.x + threadIdx.x;
  const int total = BATCH * NQ * HWP;
  if (idx >= total) return;
  int m = idx % HWP;
  int t = idx / HWP;
  int qi = t % NQ;
  int b = t / NQ;
  _Float16* dst = Qf + ((size_t)(b * NQ + qi) * HWP + m) * CH;
  if (m >= HW) {
#pragma unroll
    for (int cc = 0; cc < CH; ++cc) dst[cc] = (_Float16)0.0f;
    return;
  }
  float v[CH];
  float ss = 0.0f;
#pragma unroll 4
  for (int cc = 0; cc < CH; ++cc) {
    float x = qry[(((size_t)b * NQ + qi) * CH + cc) * HW + m];
    v[cc] = x;
    ss += x * x;
  }
  float r = 1.0f / fmaxf(sqrtf(ss), 1e-12f);
#pragma unroll 4
  for (int cc = 0; cc < CH; ++cc) dst[cc] = (_Float16)(v[cc] * r);
}

// ---------------------------------------------------------------------------
// Kernel 3: per (b,q,w) workgroup: sim = Q(448x64) x S^T(64x448) via WMMA,
// rowmax over support locations, sum over query locations -> 1 scalar.
// Tile counts arrive as RUNTIME arguments so the compiler cannot fully unroll
// the n loop / interchange loops / promote the whole B matrix into VGPRs
// (which previously caused scratch spills in the hot loop).  Two independent
// n-tiles are processed per iteration to cover the WMMA->VALU hazard.
//   mTiles = 28, nPairs = 13  (pairs cover tiles 0..25; 26 and 27 peeled,
//   27 is the only tile with padded columns to mask).
// ---------------------------------------------------------------------------
__global__ void __launch_bounds__(256)
dn4_sim(const _Float16* __restrict__ Qf, const _Float16* __restrict__ Bt,
        float* __restrict__ out, int mTiles, int nPairs) {
  __shared__ __align__(16) _Float16 ldsB[HWP * CH];  // 56 KB class tile
  __shared__ float partials[16];

  const int gid = blockIdx.x;
  const int w_i = gid % NWAY;
  const int t = gid / NWAY;
  const int q_i = t % NQ;
  const int b_i = t / NQ;

  const _Float16* Qimg = Qf + (size_t)(b_i * NQ + q_i) * HWP * CH;
  const _Float16* Bcls = Bt + (size_t)(b_i * NWAY + w_i) * HWP * CH;

  // Stage the whole support class (448x64 f16) into LDS.
  {
    const uint4* src = (const uint4*)Bcls;
    uint4* dst = (uint4*)ldsB;
#pragma unroll
    for (int i = threadIdx.x; i < (HWP * CH) / 8; i += 256) dst[i] = src[i];
  }
  __syncthreads();

  const int lane = threadIdx.x & 31;
  const int wave = threadIdx.x >> 5;
  const int laneLo = lane & 15;   // N (or M for A) index within tile
  const int laneHi = lane >> 4;   // selects contiguous 16-wide K sub-range (this code's order)

  // Per-lane base into the LDS B tile: row (laneLo), K-half (laneHi).
  const _Float16* bbase = ldsB + (size_t)laneLo * CH + laneHi * 16;

  float acc = 0.0f;  // meaningful on laneLo==0 of each half after reduction

  for (int mt = wave; mt < mTiles; mt += 8) {
    // A fragments: 16x32 f16, lane L<16 holds M=L, K=0-15;
    // lane L>=16 holds M=L-16, K=16-31 (matches contiguous B fill).
    const _Float16* arow = Qimg + (size_t)(mt * 16 + laneLo) * CH;
    v8h a00 = *(const v8h*)(arow + laneHi * 16);
    v8h a01 = *(const v8h*)(arow + 8 + laneHi * 16);
    v8h a10 = *(const v8h*)(arow + 32 + laneHi * 16);
    v8h a11 = *(const v8h*)(arow + 40 + laneHi * 16);
    v16h A0 = __builtin_shufflevector(a00, a01, 0, 1, 2, 3, 4, 5, 6, 7, 8, 9,
                                      10, 11, 12, 13, 14, 15);
    v16h A1 = __builtin_shufflevector(a10, a11, 0, 1, 2, 3, 4, 5, 6, 7, 8, 9,
                                      10, 11, 12, 13, 14, 15);

    // Prefetch the next m-tile's A row (gfx1250 global_prefetch path).
    __builtin_prefetch(arow + (size_t)8 * 16 * CH, 0, 3);

    v8f rmax;
#pragma unroll
    for (int i = 0; i < 8; ++i) rmax[i] = -__builtin_inff();

    // Hot loop: two independent n-tiles per iteration (runtime trip count).
    int nt = 0;
    const _Float16* bp = bbase;
    for (int p = 0; p < nPairs; ++p) {
      v16h B0a = *(const v16h*)(bp);                  // tile nt,   K 0..31
      v16h B1a = *(const v16h*)(bp + 32);             // tile nt,   K 32..63
      v16h B0b = *(const v16h*)(bp + 16 * CH);        // tile nt+1, K 0..31
      v16h B1b = *(const v16h*)(bp + 16 * CH + 32);   // tile nt+1, K 32..63
      v8f ca = {};
      v8f cb = {};
      ca = __builtin_amdgcn_wmma_f32_16x16x32_f16(false, A0, false, B0a,
                                                  (short)0, ca, false, false);
      cb = __builtin_amdgcn_wmma_f32_16x16x32_f16(false, A0, false, B0b,
                                                  (short)0, cb, false, false);
      ca = __builtin_amdgcn_wmma_f32_16x16x32_f16(false, A1, false, B1a,
                                                  (short)0, ca, false, false);
      cb = __builtin_amdgcn_wmma_f32_16x16x32_f16(false, A1, false, B1b,
                                                  (short)0, cb, false, false);
#pragma unroll
      for (int i = 0; i < 8; ++i) rmax[i] = fmaxf(rmax[i], fmaxf(ca[i], cb[i]));
      bp += 2 * 16 * CH;
      nt += 2;
    }
    // Peeled tile (nt == 26): all columns valid.
    {
      v16h B0 = *(const v16h*)(bp);
      v16h B1 = *(const v16h*)(bp + 32);
      v8f c0 = {};
      c0 = __builtin_amdgcn_wmma_f32_16x16x32_f16(false, A0, false, B0,
                                                  (short)0, c0, false, false);
      c0 = __builtin_amdgcn_wmma_f32_16x16x32_f16(false, A1, false, B1,
                                                  (short)0, c0, false, false);
#pragma unroll
      for (int i = 0; i < 8; ++i) rmax[i] = fmaxf(rmax[i], c0[i]);
    }
    // Peeled last tile (nt+1 == 27): mask padded support columns (N >= 441).
    {
      v16h B0 = *(const v16h*)(bp + 16 * CH);
      v16h B1 = *(const v16h*)(bp + 16 * CH + 32);
      v8f c0 = {};
      c0 = __builtin_amdgcn_wmma_f32_16x16x32_f16(false, A0, false, B0,
                                                  (short)0, c0, false, false);
      c0 = __builtin_amdgcn_wmma_f32_16x16x32_f16(false, A1, false, B1,
                                                  (short)0, c0, false, false);
      const bool colValid = ((nt + 1) * 16 + laneLo) < HW;
#pragma unroll
      for (int i = 0; i < 8; ++i) {
        float v = colValid ? c0[i] : -__builtin_inff();
        rmax[i] = fmaxf(rmax[i], v);
      }
    }

    // Max over N: reduce across the 16 lanes of each half (C layout puts
    // M=i (+8 for upper half) at VGPR i, N = lane%16).
#pragma unroll
    for (int mask = 1; mask < 16; mask <<= 1) {
#pragma unroll
      for (int i = 0; i < 8; ++i) {
        float o = __shfl_xor(rmax[i], mask, 32);
        rmax[i] = fmaxf(rmax[i], o);
      }
    }
    float p = 0.0f;
#pragma unroll
    for (int i = 0; i < 8; ++i) p += rmax[i];  // sum of 8 row maxima
    acc += p;
  }

  if (laneLo == 0) partials[wave * 2 + laneHi] = acc;
  __syncthreads();
  if (threadIdx.x == 0) {
    float s = 0.0f;
#pragma unroll
    for (int i = 0; i < 16; ++i) s += partials[i];
    out[(size_t)(b_i * NQ + q_i) * NWAY + w_i] = s;
  }
}

// ---------------------------------------------------------------------------
extern "C" void kernel_launch(void* const* d_in, const int* in_sizes, int n_in,
                              void* d_out, int out_size, void* d_ws,
                              size_t ws_size, hipStream_t stream) {
  const float* support_xf = (const float*)d_in[0];
  const float* query_xf = (const float*)d_in[2];
  float* logits = (float*)d_out;

  // Workspace layout: Qf (150*448*64 f16 = 8.6 MB), then Bt (10*448*64 f16).
  _Float16* Qf = (_Float16*)d_ws;
  _Float16* Bt = Qf + (size_t)BATCH * NQ * HWP * CH;

  {
    int total = BATCH * NWAY * HWP;
    dn4_prep_support<<<(total + 255) / 256, 256, 0, stream>>>(support_xf, Bt);
  }
  {
    int total = BATCH * NQ * HWP;
    dn4_prep_query<<<(total + 255) / 256, 256, 0, stream>>>(query_xf, Qf);
  }
  {
    int grid = BATCH * NQ * NWAY;  // 750 workgroups, one per (b, q, class)
    // Runtime tile counts (28 m-tiles, 13 n-tile pairs) keep the compiler
    // from fully unrolling / hoisting B into registers.
    dn4_sim<<<grid, 256, 0, stream>>>(Qf, Bt, logits, NT, (NT - 2) / 2);
  }
}
